// GPUReverb_8323646620199
// MI455X (gfx1250) — matrix operand order repaired and checked by reading, not compiled, not run on hardware
//
#include <hip/hip_runtime.h>
#include <hip/hip_bf16.h>

// ---------------- problem constants ----------------
#define B_      64
#define T_      160000
#define R_      16000
#define DBEF    16
#define DAFT    80
#define EARLY   96

// WMMA conv tiling: out[t0 + m + 16n] += sum_k A[m,k]*B[k,n]
//   A[m,k] = rx[DCONST - (t0+m+sb-base) + k]   (rx = reversed, padded noisy, f16)
//   B[k,n] = h16[HOFF + base + 16n + k]        (h16 = padded rir, f16, LDS-staged)
// base = BASE_MIN + 32*it, it in [0, NITER)
#define HOFF     256
#define HLEN     16544            // 256 front pad + 16000 + 288 back pad (halfs)
#define HCHUNKS  2068             // HLEN*2 bytes / 16 = b128 chunks
#define BASE_MIN (-256)
#define NITER    508              // last base = -256 + 32*507 = 15968
#define DCONST   176238
#define RXLEN    192256
#define TILES_PB 625              // 256-output tiles per batch
#define BLKS_PB  79               // 79 blocks * 8 waves = 632 >= 625

typedef __attribute__((ext_vector_type(16))) _Float16 v16h;
typedef __attribute__((ext_vector_type(8)))  _Float16 v8h;
typedef __attribute__((ext_vector_type(8)))  float    v8f;

// ---------------- K0: per-batch argmax|rir|, init peak ----------------
__global__ __launch_bounds__(256) void k_argmax(const float* __restrict__ rir,
                                                int* __restrict__ start,
                                                int* __restrict__ endv,
                                                unsigned* __restrict__ peak) {
    int b = blockIdx.x;
    __shared__ float sv[256];
    __shared__ int   si[256];
    float best = -1.0f; int bidx = 0;
    for (int i = threadIdx.x; i < R_; i += 256) {
        float v = fabsf(rir[(size_t)b * R_ + i]);
        if (v > best) { best = v; bidx = i; }
    }
    sv[threadIdx.x] = best; si[threadIdx.x] = bidx;
    __syncthreads();
    for (int s = 128; s > 0; s >>= 1) {
        if (threadIdx.x < s) {
            float ov = sv[threadIdx.x + s]; int oi = si[threadIdx.x + s];
            if (ov > sv[threadIdx.x] ||
                (ov == sv[threadIdx.x] && oi < si[threadIdx.x])) {
                sv[threadIdx.x] = ov; si[threadIdx.x] = oi;
            }
        }
        __syncthreads();
    }
    if (threadIdx.x == 0) {
        int d = si[0];
        start[b] = max(0, d - DBEF);
        endv[b]  = d + DAFT;
        peak[b]  = 0u;                 // ws is poisoned; must init
    }
}

// ---------------- K1: pack zero-padded f16 filter ----------------
__global__ __launch_bounds__(256) void k_packh(const float* __restrict__ rir,
                                               _Float16* __restrict__ h16) {
    int idx = blockIdx.x * 256 + threadIdx.x;
    if (idx >= B_ * HLEN) return;
    int b = idx / HLEN, i = idx % HLEN;
    int j = i - HOFF;
    float v = (j >= 0 && j < R_) ? rir[(size_t)b * R_ + j] : 0.0f;
    h16[idx] = (_Float16)v;
}

// ---------------- K2: pack reversed zero-padded f16 signal ----------------
__global__ __launch_bounds__(256) void k_packx(const float* __restrict__ noisy,
                                               _Float16* __restrict__ rx) {
    int idx = blockIdx.x * 256 + threadIdx.x;
    if (idx >= B_ * RXLEN) return;
    int b = idx / RXLEN, i = idx % RXLEN;
    int t = DCONST - i;
    float v = (t >= 0 && t < T_) ? noisy[(size_t)b * T_ + t] : 0.0f;
    rx[idx] = (_Float16)v;
}

// ---------------- K3: full reverb via WMMA f16 (f32 accumulate) ----------------
// one block = one batch's 8-tile group; filter staged to LDS with async copies.
// All conv branches are wave-uniform, so EXEC is all-ones whenever WMMA executes.
__global__ __launch_bounds__(256) void k_conv_wmma(const _Float16* __restrict__ rx,
                                                   const _Float16* __restrict__ h16,
                                                   const int* __restrict__ start,
                                                   unsigned* __restrict__ peak,
                                                   float* __restrict__ out) {
    __shared__ __align__(32) _Float16 sh_h[HLEN];

    const int batch = blockIdx.x / BLKS_PB;
    const int blk   = blockIdx.x % BLKS_PB;
    const int lane  = threadIdx.x & 31;
    const int tile  = (blk << 3) + (threadIdx.x >> 5);

    // ---- stage this batch's packed filter into LDS (gfx1250 async-to-LDS) ----
    {
        const void* hbase = (const void*)(h16 + (size_t)batch * HLEN);
        unsigned shbase = (unsigned)(uintptr_t)(&sh_h[0]);  // low 32 bits = LDS offset
        for (int c = threadIdx.x; c < HCHUNKS; c += 256) {
            unsigned off = (unsigned)c * 16u;
            asm volatile("global_load_async_to_lds_b128 %0, %1, %2"
                         :
                         : "v"(shbase + off), "v"(off), "s"(hbase)
                         : "memory");
        }
        asm volatile("s_wait_asynccnt 0x0" ::: "memory");
    }
    __syncthreads();

    if (tile < TILES_PB) {                 // wave-uniform guard (7 tail waves idle)
        const int t0 = tile << 8;
        const int sb = start[batch];

        const _Float16* rxb = rx + (size_t)batch * RXLEN;
        const int m  = lane & 15;          // A row / B column
        const int hi = lane >> 4;          // half-wave selector
        // A (16x32 f16): lanes 0-15 hold K 0..7 & 16..23; lanes 16-31 K 8..15 & 24..31
        const _Float16* aptr = rxb + (DCONST - t0 - sb + BASE_MIN - m + (hi << 3));
        // B (32x16 f16): column m from LDS; lanes 0-15 hold K 0..7 & 16..23; lanes 16-31 K 8..15 & 24..31
        const _Float16* bptr = sh_h + (HOFF + BASE_MIN) + (m << 4) + (hi << 3);

        v8f acc = {};
#pragma unroll 4
        for (int it = 0; it < NITER; ++it) {
            v8h a_lo, a_hi;
            __builtin_memcpy(&a_lo, aptr,      16);   // 2B-aligned Toeplitz reads
            __builtin_memcpy(&a_hi, aptr + 16, 16);
            v8h b_lo = *(const v8h*)(bptr);           // ds_load_b128
            v8h b_hi = *(const v8h*)(bptr + 16);
            v16h a  = __builtin_shufflevector(a_lo, a_hi,
                        0,1,2,3,4,5,6,7,8,9,10,11,12,13,14,15);
            v16h bm = __builtin_shufflevector(b_lo, b_hi,
                        0,1,2,3,4,5,6,7,8,9,10,11,12,13,14,15);
            acc = __builtin_amdgcn_wmma_f32_16x16x32_f16(
                /*neg_a=*/false, a, /*neg_b=*/false, bm,
                /*c_mod=*/(short)0, acc, /*reuse_a=*/false, /*reuse_b=*/false);
            __builtin_prefetch(aptr + 1024, 0, 0);    // global_prefetch, A stream
            aptr += 32;
            bptr += 32;
        }

        // D layout: vgpr r, lane -> out[t0 + r + 8*hi + 16*m] (8 consecutive floats)
        size_t ob = (size_t)batch * T_ + t0 + (hi << 3) + (m << 4);
        *(v8f*)(out + ob) = acc;

        // per-batch peak of |rev_full| (non-negative: uint-bit atomicMax is exact)
        float pm = 0.0f;
#pragma unroll
        for (int r = 0; r < 8; ++r) pm = fmaxf(pm, fabsf(acc[r]));
        for (int off = 16; off > 0; off >>= 1)
            pm = fmaxf(pm, __shfl_xor(pm, off, 32));
        if (lane == 0) atomicMax(peak + batch, __float_as_uint(pm));
    }
}

// ---------------- K4: 96-tap early-reflection FIR (f32, LDS-staged) ----------------
__global__ __launch_bounds__(256) void k_early(const float* __restrict__ speech,
                                               const float* __restrict__ rir,
                                               const int* __restrict__ start,
                                               const int* __restrict__ endv,
                                               float* __restrict__ out_early) {
    __shared__ float tap[EARLY];
    __shared__ float xs[256 + EARLY];
    const int b  = blockIdx.x / TILES_PB;
    const int t0 = (blockIdx.x % TILES_PB) << 8;
    const int tx = threadIdx.x;

    if (tx < EARLY) {
        int s = start[b], e = endv[b];
        int pos = s + tx;
        tap[tx] = (pos < e && pos < R_) ? rir[(size_t)b * R_ + pos] : 0.0f;
    }
    for (int i = tx; i < 256 + EARLY - 1; i += 256) {
        int t = t0 - (EARLY - 1) + i;
        xs[i] = (t >= 0 && t < T_) ? speech[(size_t)b * T_ + t] : 0.0f;
    }
    __syncthreads();

    float s = 0.0f;
#pragma unroll
    for (int e = 0; e < EARLY; ++e)
        s = fmaf(xs[tx + (EARLY - 1) - e], tap[e], s);
    out_early[(size_t)b * T_ + t0 + tx] = s;
}

// ---------------- K5: apply peak normalization to both outputs ----------------
__global__ __launch_bounds__(256) void k_scale(const unsigned* __restrict__ peak,
                                               float* __restrict__ out) {
    int idx = blockIdx.x * 256 + threadIdx.x;
    if (idx >= B_ * T_) return;
    int b = idx / T_;
    float p = __uint_as_float(peak[b]);
    float sc = (p > 0.0f) ? 0.5f / p : 1.0f;
    out[idx] *= sc;
    out[(size_t)B_ * T_ + idx] *= sc;
}

// ---------------- launcher ----------------
extern "C" void kernel_launch(void* const* d_in, const int* in_sizes, int n_in,
                              void* d_out, int out_size, void* d_ws, size_t ws_size,
                              hipStream_t stream) {
    const float* speech = (const float*)d_in[0];
    const float* noisy  = (const float*)d_in[1];
    const float* rir    = (const float*)d_in[2];
    float* out = (float*)d_out;

    char* ws = (char*)d_ws;
    int*      start = (int*)(ws + 0);
    int*      endv  = (int*)(ws + 256);
    unsigned* peak  = (unsigned*)(ws + 512);
    _Float16* h16   = (_Float16*)(ws + 1024);
    _Float16* rx    = (_Float16*)(ws + 1024 + (size_t)B_ * HLEN * 2);

    k_argmax<<<B_, 256, 0, stream>>>(rir, start, endv, peak);
    k_packh <<<(B_ * HLEN  + 255) / 256, 256, 0, stream>>>(rir, h16);
    k_packx <<<(B_ * RXLEN + 255) / 256, 256, 0, stream>>>(noisy, rx);
    k_conv_wmma<<<B_ * BLKS_PB, 256, 0, stream>>>(rx, h16, start, peak, out);
    k_early <<<B_ * TILES_PB, 256, 0, stream>>>(speech, rir, start, endv,
                                                out + (size_t)B_ * T_);
    k_scale <<<(B_ * T_ + 255) / 256, 256, 0, stream>>>(peak, out);
}
